// CasualMultiHeadSelfAttention_7327214207591
// MI455X (gfx1250) — matrix operand order repaired and checked by reading, not compiled, not run on hardware
//
#include <hip/hip_runtime.h>
#include <hip/hip_bf16.h>
#include <math.h>

typedef _Float16 v16h __attribute__((ext_vector_type(16)));
typedef _Float16 v8h  __attribute__((ext_vector_type(8)));
typedef float    v8f  __attribute__((ext_vector_type(8)));
typedef int      v4i  __attribute__((ext_vector_type(4)));

#if defined(__has_builtin)
# if __has_builtin(__builtin_amdgcn_global_load_async_to_lds_b128)
#  define HAVE_ASYNC_LDS 1
# endif
#endif
#ifndef HAVE_ASYNC_LDS
# define HAVE_ASYNC_LDS 0
#endif

#if HAVE_ASYNC_LDS
#define ASYNC_CP16(gp, lp)                                                   \
  __builtin_amdgcn_global_load_async_to_lds_b128(                            \
      (__attribute__((address_space(1))) v4i*)(gp),                          \
      (__attribute__((address_space(3))) v4i*)(lp), 0, 0)
#endif

static __device__ __forceinline__ v16h combine16(v8h lo, v8h hi){
  v16h r;
  #pragma unroll
  for (int i = 0; i < 8; ++i){ r[i] = lo[i]; r[8+i] = hi[i]; }
  return r;
}
static __device__ __forceinline__ v8h ldg8h(const _Float16* p){ return *(const v8h*)p; }

// ---------------- fp32 -> f16 convert ----------------
__global__ void cvt_f32_to_f16(const float* __restrict__ in, _Float16* __restrict__ out, int n){
  int i = blockIdx.x * blockDim.x + threadIdx.x;
  int stride = gridDim.x * blockDim.x;
  for (; i < n; i += stride) out[i] = (_Float16)in[i];
}

// ---------------- NT GEMM: C[M,N] = A[M,K] x B[N,K]^T (f16 in, f32 accum) ----------------
// 128x128 block tile, 8 waves (4 along M x 2 along N), double-buffered 32-wide K slabs,
// async global->LDS staging when available.
template<int N, int K, bool F32OUT>
__global__ __launch_bounds__(256) void gemm_nt_wmma(const _Float16* __restrict__ A,
                                                    const _Float16* __restrict__ B,
                                                    void* __restrict__ Cp){
  __shared__ __align__(16) _Float16 As[2][128 * 40];
  __shared__ __align__(16) _Float16 Bs[2][128 * 40];
  const int t    = threadIdx.x;
  const int wave = t >> 5, lane = t & 31;
  const int half = lane >> 4, l16 = lane & 15;
  const int bm = blockIdx.y * 128, bn = blockIdx.x * 128;
  const int wm = (wave & 3) * 32;   // 32 rows per wave
  const int wn = (wave >> 2) * 64;  // 64 cols per wave

  const v8f zero = {0.f,0.f,0.f,0.f,0.f,0.f,0.f,0.f};
  v8f acc[2][4];
  #pragma unroll
  for (int i = 0; i < 2; ++i)
    #pragma unroll
    for (int j = 0; j < 4; ++j) acc[i][j] = zero;

  // staging: 2 threads per 32-f16 row, 32B per thread per matrix
  const int row = t >> 1, hf = t & 1;
  const _Float16* gA = A + (size_t)(bm + row) * K + hf * 16;
  const _Float16* gB = B + (size_t)(bn + row) * K + hf * 16;
  const int ldsOff = row * 40 + hf * 16;

  auto stage = [&](int buf, int k0){
#if HAVE_ASYNC_LDS
    ASYNC_CP16(gA + k0,     &As[buf][ldsOff]);
    ASYNC_CP16(gA + k0 + 8, &As[buf][ldsOff + 8]);
    ASYNC_CP16(gB + k0,     &Bs[buf][ldsOff]);
    ASYNC_CP16(gB + k0 + 8, &Bs[buf][ldsOff + 8]);
#else
    v8h a0 = ldg8h(gA + k0), a1 = ldg8h(gA + k0 + 8);
    v8h b0 = ldg8h(gB + k0), b1 = ldg8h(gB + k0 + 8);
    *(v8h*)&As[buf][ldsOff]     = a0;
    *(v8h*)&As[buf][ldsOff + 8] = a1;
    *(v8h*)&Bs[buf][ldsOff]     = b0;
    *(v8h*)&Bs[buf][ldsOff + 8] = b1;
#endif
  };

  stage(0, 0);
  for (int k0 = 0; k0 < K; k0 += 32){
    const int cur = (k0 >> 5) & 1;
#if HAVE_ASYNC_LDS
    asm volatile("s_wait_asynccnt 0x0" ::: "memory");
#endif
    __syncthreads();                       // staged slab visible to all waves
    if (k0 + 32 < K) stage(cur ^ 1, k0 + 32);   // overlap next slab with compute

    v16h aF[2], bF[4];
    #pragma unroll
    for (int mt = 0; mt < 2; ++mt){
      const int r0 = wm + mt * 16 + l16;   // A frag: lane = M row, K pattern {h*8.., 16+h*8..}
      aF[mt] = combine16(*(const v8h*)&As[cur][r0 * 40 + half * 8],
                         *(const v8h*)&As[cur][r0 * 40 + 16 + half * 8]);
    }
    #pragma unroll
    for (int nt = 0; nt < 4; ++nt){
      const int r0 = wn + nt * 16 + l16;   // B frag: lane = N col, K pattern {h*8.., 16+h*8..}
      bF[nt] = combine16(*(const v8h*)&Bs[cur][r0 * 40 + half * 8],
                         *(const v8h*)&Bs[cur][r0 * 40 + 16 + half * 8]);
    }
    #pragma unroll
    for (int mt = 0; mt < 2; ++mt)
      #pragma unroll
      for (int nt = 0; nt < 4; ++nt)
        acc[mt][nt] = __builtin_amdgcn_wmma_f32_16x16x32_f16(
            false, aF[mt], false, bF[nt], (short)0, acc[mt][nt], false, false);
  }

  // epilogue: one base address per accumulator tile, immediate row offsets (r*N folds into IOFFSET)
  #pragma unroll
  for (int mt = 0; mt < 2; ++mt)
    #pragma unroll
    for (int nt = 0; nt < 4; ++nt){
      const int m0 = bm + wm + mt * 16 + 8 * half;  // C layout: row = r + 8*(lane/16)
      const int n  = bn + wn + nt * 16 + l16;       //           col = lane%16
      if constexpr (F32OUT){
        float* base = (float*)Cp + (size_t)m0 * N + n;
        #pragma unroll
        for (int r = 0; r < 8; ++r) base[(size_t)r * N] = acc[mt][nt][r];
      } else {
        _Float16* base = (_Float16*)Cp + (size_t)m0 * N + n;
        #pragma unroll
        for (int r = 0; r < 8; ++r) base[(size_t)r * N] = (_Float16)acc[mt][nt][r];
      }
    }
}

// ---------------- RoPE (in place on f16 [B,S,H,64], interleaved pairs) ----------------
__global__ void rope_f16(_Float16* __restrict__ qk, int npairs){
  const int idx = blockIdx.x * blockDim.x + threadIdx.x;
  if (idx >= npairs) return;
  const int i = idx & 31;          // pair index within head (d_k/2 = 32)
  int rest = idx >> 5;
  const int h = rest & 15; rest >>= 4;
  const int s = rest & 2047;
  const int b = rest >> 11;
  const float inv = __expf(-(2.f * (float)i / 64.f) * 9.210340371976184f); // 10000^(-2i/64)
  const float ang = (float)s * inv;
  float sn, cs; __sincosf(ang, &sn, &cs);
  const size_t off = (((size_t)(b * 2048 + s) * 16 + h) * 64) + 2 * i;
  const float e = (float)qk[off], o = (float)qk[off + 1];
  qk[off]     = (_Float16)(e * cs - o * sn);
  qk[off + 1] = (_Float16)(o * cs + e * sn);
}

// ---------------- V transpose: v[b,s,h,d] -> vt[b,h,d,s] ----------------
__global__ void v_transpose(const _Float16* __restrict__ v, _Float16* __restrict__ vt, int n){
  const int idx = blockIdx.x * blockDim.x + threadIdx.x;
  if (idx >= n) return;
  const int s = idx & 2047;
  int r = idx >> 11;
  const int d = r & 63; r >>= 6;
  const int h = r & 15;
  const int b = r >> 4;
  vt[idx] = v[(((size_t)(b * 2048 + s) * 16 + h) * 64) + d];
}

// ---------------- Flash attention: 1 wave = 16 query rows, 4 waves/block ----------------
__global__ __launch_bounds__(128) void flash_attn_wmma(const _Float16* __restrict__ qh,
                                                       const _Float16* __restrict__ kh,
                                                       const _Float16* __restrict__ vth,
                                                       _Float16* __restrict__ oh){
  __shared__ __align__(16) _Float16 Pb[4][16 * 40];   // per-wave P staging (C-layout -> A-layout)
  const int wave = threadIdx.x >> 5, lane = threadIdx.x & 31;
  const int half = lane >> 4, l16 = lane & 15;
  const int qblk = blockIdx.x & 31;
  const int head = (blockIdx.x >> 5) & 15;
  const int b    = blockIdx.x >> 9;
  const int q0   = qblk * 64 + wave * 16;

  // Q fragments (A operand, 16x64 split into two 16x32 K-blocks)
  const _Float16* qrow = qh + (((size_t)(b * 2048 + q0 + l16) * 16 + head) * 64);
  v16h Qf[2];
  #pragma unroll
  for (int c = 0; c < 2; ++c)
    Qf[c] = combine16(ldg8h(qrow + c * 32 + half * 8), ldg8h(qrow + c * 32 + 16 + half * 8));

  const v8f zero = {0.f,0.f,0.f,0.f,0.f,0.f,0.f,0.f};
  v8f O[4];
  #pragma unroll
  for (int nb = 0; nb < 4; ++nb) O[nb] = zero;
  float mrow[8], lrow[8];
  #pragma unroll
  for (int r = 0; r < 8; ++r){ mrow[r] = -1e30f; lrow[r] = 0.f; }

  const int nIter = (q0 + 47) >> 5;       // ceil((q0+16)/32) 32-key steps
  for (int it = 0; it < nIter; ++it){
    const int j0 = it * 32;

    // K fragments (B operand for scores): two 16-key tiles x two 32-K dk blocks
    v16h Kf[2][2];
    #pragma unroll
    for (int tt = 0; tt < 2; ++tt){
      const _Float16* krow = kh + (((size_t)(b * 2048 + j0 + tt * 16 + l16) * 16 + head) * 64);
      #pragma unroll
      for (int c = 0; c < 2; ++c)
        Kf[tt][c] = combine16(ldg8h(krow + c * 32 + half * 8),
                              ldg8h(krow + c * 32 + 16 + half * 8));
    }
    // V fragments (B operand for PV): K pattern {h*8.., 16+h*8..} over 32 keys in vt rows
    v16h Vf[4];
    #pragma unroll
    for (int nb = 0; nb < 4; ++nb){
      const _Float16* vrow = vth + (((size_t)(b * 16 + head) * 64 + nb * 16 + l16) * 2048) + j0;
      Vf[nb] = combine16(ldg8h(vrow + half * 8), ldg8h(vrow + 16 + half * 8));
    }

    // scores: two 16x16 tiles, each two K=32 WMMAs over d_k=64
    v8f Sc[2];
    #pragma unroll
    for (int tt = 0; tt < 2; ++tt){
      Sc[tt] = __builtin_amdgcn_wmma_f32_16x16x32_f16(false, Qf[0], false, Kf[tt][0],
                                                      (short)0, zero, false, false);
      Sc[tt] = __builtin_amdgcn_wmma_f32_16x16x32_f16(false, Qf[1], false, Kf[tt][1],
                                                      (short)0, Sc[tt], false, false);
    }

    // causal mask + scale + online softmax (rows live across the 16-lane half)
    float p0[8], p1[8];
    #pragma unroll
    for (int r = 0; r < 8; ++r){
      const int qi = q0 + r + 8 * half;
      float a  = Sc[0][r] * 0.125f; if (j0 + l16      > qi) a  = -1e30f;
      float c2 = Sc[1][r] * 0.125f; if (j0 + 16 + l16 > qi) c2 = -1e30f;
      float mx = fmaxf(a, c2);
      #pragma unroll
      for (int off = 8; off >= 1; off >>= 1) mx = fmaxf(mx, __shfl_xor(mx, off, 32));
      const float mnew = fmaxf(mrow[r], mx);
      const float corr = __expf(mrow[r] - mnew);
      const float e0 = __expf(a - mnew), e1 = __expf(c2 - mnew);
      float rs = e0 + e1;
      #pragma unroll
      for (int off = 8; off >= 1; off >>= 1) rs += __shfl_xor(rs, off, 32);
      lrow[r] = lrow[r] * corr + rs;
      mrow[r] = mnew;
      #pragma unroll
      for (int nb = 0; nb < 4; ++nb) O[nb][r] *= corr;
      p0[r] = e0; p1[r] = e1;
    }

    // repack P from C-layout to A-layout through wave-private LDS
    asm volatile("s_wait_dscnt 0" ::: "memory");
    #pragma unroll
    for (int r = 0; r < 8; ++r){
      Pb[wave][(r + 8 * half) * 40 + l16]      = (_Float16)p0[r];
      Pb[wave][(r + 8 * half) * 40 + 16 + l16] = (_Float16)p1[r];
    }
    asm volatile("s_wait_dscnt 0" ::: "memory");
    v16h Pf = combine16(*(const v8h*)&Pb[wave][l16 * 40 + half * 8],
                        *(const v8h*)&Pb[wave][l16 * 40 + 16 + half * 8]);

    #pragma unroll
    for (int nb = 0; nb < 4; ++nb)
      O[nb] = __builtin_amdgcn_wmma_f32_16x16x32_f16(false, Pf, false, Vf[nb],
                                                     (short)0, O[nb], false, false);
  }

  // finalize: divide by l, write f16 attn output [b,s,h*64+d]
  #pragma unroll
  for (int r = 0; r < 8; ++r){
    const float inv = 1.f / lrow[r];
    const int m = q0 + r + 8 * half;
    #pragma unroll
    for (int nb = 0; nb < 4; ++nb)
      oh[((size_t)(b * 2048 + m) * 1024) + head * 64 + nb * 16 + l16] =
          (_Float16)(O[nb][r] * inv);
  }
}

extern "C" void kernel_launch(void* const* d_in, const int* in_sizes, int n_in,
                              void* d_out, int out_size, void* d_ws, size_t ws_size,
                              hipStream_t stream) {
  const float* x  = (const float*)d_in[0];
  const float* wq = (const float*)d_in[1];
  const float* wk = (const float*)d_in[2];
  const float* wv = (const float*)d_in[3];
  const float* wo = (const float*)d_in[4];

  _Float16* W = (_Float16*)d_ws;
  const size_t MB = 1024u * 1024u;       // elements, not bytes
  _Float16* xh  = W;                     // 4M f16
  _Float16* wqh = W + 4  * MB;           // 1M each
  _Float16* wkh = W + 5  * MB;
  _Float16* wvh = W + 6  * MB;
  _Float16* woh = W + 7  * MB;
  _Float16* qhp = W + 8  * MB;           // 4M each
  _Float16* khp = W + 12 * MB;
  _Float16* vhp = W + 16 * MB;
  _Float16* vtp = W + 20 * MB;
  _Float16* ohp = W + 24 * MB;           // total 28M f16 = 56MB

  cvt_f32_to_f16<<<4096, 256, 0, stream>>>(x,  xh,  4 * 1024 * 1024);
  cvt_f32_to_f16<<<2048, 256, 0, stream>>>(wq, wqh, 1024 * 1024);
  cvt_f32_to_f16<<<2048, 256, 0, stream>>>(wk, wkh, 1024 * 1024);
  cvt_f32_to_f16<<<2048, 256, 0, stream>>>(wv, wvh, 1024 * 1024);
  cvt_f32_to_f16<<<2048, 256, 0, stream>>>(wo, woh, 1024 * 1024);

  dim3 gg(1024 / 128, 4096 / 128);
  gemm_nt_wmma<1024, 1024, false><<<gg, 256, 0, stream>>>(xh, wqh, qhp);
  gemm_nt_wmma<1024, 1024, false><<<gg, 256, 0, stream>>>(xh, wkh, khp);
  gemm_nt_wmma<1024, 1024, false><<<gg, 256, 0, stream>>>(xh, wvh, vhp);

  const int npairs = 2 * 2048 * 16 * 32;
  rope_f16<<<(npairs + 255) / 256, 256, 0, stream>>>(qhp, npairs);
  rope_f16<<<(npairs + 255) / 256, 256, 0, stream>>>(khp, npairs);

  const int nvt = 2 * 16 * 64 * 2048;
  v_transpose<<<(nvt + 255) / 256, 256, 0, stream>>>(vhp, vtp, nvt);

  flash_attn_wmma<<<2 * 16 * 32, 128, 0, stream>>>(qhp, khp, vtp, ohp);

  gemm_nt_wmma<1024, 1024, true><<<gg, 256, 0, stream>>>(ohp, woh, d_out);
}
